// AttentionUtil_14551349198869
// MI455X (gfx1250) — matrix operand order repaired and checked by reading, not compiled, not run on hardware
//
#include <hip/hip_runtime.h>
#include <hip/hip_bf16.h>

typedef __attribute__((ext_vector_type(16))) _Float16 v16h;
typedef __attribute__((ext_vector_type(8)))  float    v8f;

#define B_SZ 16
#define N_SZ 2048
#define D_SZ 64
#define BM   128  // query rows per workgroup (8 waves x 16 rows)
#define BN   64   // key rows per iteration
#define NWAVE 8
#define NTHREADS (NWAVE * 32)
#define NJ   (N_SZ / BN)

// ---- fused DPP16 butterfly reductions across a 16-lane row (1 instr/step) ----
__device__ __forceinline__ float red_max16(float x) {
  asm("v_max_num_f32_dpp %0, %0, %0 quad_perm:[1,0,3,2] row_mask:0xf bank_mask:0xf bound_ctrl:1" : "+v"(x));
  asm("v_max_num_f32_dpp %0, %0, %0 quad_perm:[2,3,0,1] row_mask:0xf bank_mask:0xf bound_ctrl:1" : "+v"(x));
  asm("v_max_num_f32_dpp %0, %0, %0 row_half_mirror row_mask:0xf bank_mask:0xf bound_ctrl:1" : "+v"(x));
  asm("v_max_num_f32_dpp %0, %0, %0 row_mirror row_mask:0xf bank_mask:0xf bound_ctrl:1" : "+v"(x));
  return x;
}
__device__ __forceinline__ float red_sum16(float x) {
  asm("v_add_f32_dpp %0, %0, %0 quad_perm:[1,0,3,2] row_mask:0xf bank_mask:0xf bound_ctrl:1" : "+v"(x));
  asm("v_add_f32_dpp %0, %0, %0 quad_perm:[2,3,0,1] row_mask:0xf bank_mask:0xf bound_ctrl:1" : "+v"(x));
  asm("v_add_f32_dpp %0, %0, %0 row_half_mirror row_mask:0xf bank_mask:0xf bound_ctrl:1" : "+v"(x));
  asm("v_add_f32_dpp %0, %0, %0 row_mirror row_mask:0xf bank_mask:0xf bound_ctrl:1" : "+v"(x));
  return x;
}

union H4 {                      // 4 packed f16 = one 8-byte LDS store
  _Float16 h[4];
  uint2 u;
};

__global__ __launch_bounds__(NTHREADS) void
fa_wmma_kernel(const float* __restrict__ Q,
               const float* __restrict__ K,
               const float* __restrict__ V,
               float* __restrict__ O) {
  __shared__ _Float16 sQ[BM * D_SZ];          // row-major [m][d], pre-scaled (16 KB)
  __shared__ _Float16 sK[BN * D_SZ];          // row-major [n][d] (serves as B = K^T)
  __shared__ _Float16 sVt[D_SZ * BN];         // transposed [dv][k] for contiguous B frags
  __shared__ _Float16 sP[NWAVE][16 * BN];     // per-wave P strip for C->A reshape

  const int tid  = threadIdx.x;
  const int wave = tid >> 5;
  const int lane = tid & 31;
  const int hi   = lane >> 4;    // half-wave select (0/1)
  const int lq   = lane & 15;    // lane within half (= N column / M row index)

  const int nqblk = N_SZ / BM;
  const int batch = blockIdx.x / nqblk;
  const int qblk  = blockIdx.x % nqblk;

  const float* Qg = Q + ((size_t)batch * N_SZ + (size_t)qblk * BM) * D_SZ;
  const float* Kg = K + (size_t)batch * N_SZ * D_SZ;
  const float* Vg = V + (size_t)batch * N_SZ * D_SZ;

  const int k0  = (tid >> 4) * 4;    // V transpose: key-row block
  const int dv0 = (tid & 15) * 4;    // V transpose: value-column block

  // ---- prefetch K/V tile 0 into registers (overlaps with Q staging) ----
  float4 kr[4], vr[4];
  #pragma unroll
  for (int i = 0; i < 4; ++i)
    kr[i] = ((const float4*)Kg)[tid * 4 + i];
  #pragma unroll
  for (int i = 0; i < 4; ++i)
    vr[i] = *(const float4*)(Vg + (size_t)(k0 + i) * D_SZ + dv0);

  // ---- stage Q tile; fold 1/sqrt(D) * log2(e) so softmax runs in exp2 domain ----
  {
    const float qscale = 0.125f * 1.44269504088896341f;
    const float4* src = (const float4*)Qg;
    #pragma unroll
    for (int i = 0; i < 8; ++i) {
      int g = tid * 8 + i;               // 2048 float4 total (128x64)
      float4 v = src[g];
      H4 p;
      p.h[0] = (_Float16)(v.x * qscale);
      p.h[1] = (_Float16)(v.y * qscale);
      p.h[2] = (_Float16)(v.z * qscale);
      p.h[3] = (_Float16)(v.w * qscale);
      *(uint2*)&sQ[g * 4] = p.u;         // one ds_store_b64
    }
  }
  __syncthreads();

  // ---- Q A-fragments (two 16x32 k-chunks over D=64), contiguous k order (16*hi + e) ----
  v16h aQ[2];
  {
    const _Float16* row = &sQ[(wave * 16 + lq) * D_SZ];
    #pragma unroll
    for (int kc = 0; kc < 2; ++kc) {
      const int kb = kc * 32;
      #pragma unroll
      for (int e = 0; e < 8; ++e) aQ[kc][e]     = row[kb + hi * 16 + e];
      #pragma unroll
      for (int e = 0; e < 8; ++e) aQ[kc][8 + e] = row[kb + hi * 16 + 8 + e];
    }
  }

  // softmax state: row m = r + 8*hi, replicated across the 16 lanes of a half
  float mrow[8], lrow[8];
  v8f acc[4];
  const v8f vzero = {};
  #pragma unroll
  for (int r = 0; r < 8; ++r) { mrow[r] = -__builtin_inff(); lrow[r] = 0.0f; }
  #pragma unroll
  for (int t = 0; t < 4; ++t) acc[t] = vzero;

  for (int j = 0; j < NJ; ++j) {
    __syncthreads();   // previous iteration's readers done with sK/sVt

    // ---- stage prefetched K (row-major) ----
    #pragma unroll
    for (int i = 0; i < 4; ++i) {
      H4 p;
      p.h[0] = (_Float16)kr[i].x;
      p.h[1] = (_Float16)kr[i].y;
      p.h[2] = (_Float16)kr[i].z;
      p.h[3] = (_Float16)kr[i].w;
      *(uint2*)&sK[(tid * 4 + i) * 4] = p.u;
    }
    // ---- stage prefetched V via 4x4 register-block transpose ----
    {
      H4 col[4];
      #pragma unroll
      for (int i = 0; i < 4; ++i) {
        col[0].h[i] = (_Float16)vr[i].x;
        col[1].h[i] = (_Float16)vr[i].y;
        col[2].h[i] = (_Float16)vr[i].z;
        col[3].h[i] = (_Float16)vr[i].w;
      }
      #pragma unroll
      for (int c = 0; c < 4; ++c)
        *(uint2*)&sVt[(dv0 + c) * BN + k0] = col[c].u;   // ds_store_b64
    }
    __syncthreads();

    // ---- issue global prefetch for tile j+1 (overlaps with compute below) ----
    {
      const int jn = (j + 1 < NJ) ? j + 1 : j;   // clamp: branchless tail
      const float* kb = Kg + (size_t)jn * BN * D_SZ;
      const float* vb = Vg + (size_t)jn * BN * D_SZ;
      #pragma unroll
      for (int i = 0; i < 4; ++i)
        kr[i] = ((const float4*)kb)[tid * 4 + i];
      #pragma unroll
      for (int i = 0; i < 4; ++i)
        vr[i] = *(const float4*)(vb + (size_t)(k0 + i) * D_SZ + dv0);
    }

    // ---- S = Q K^T : 4 n-tiles x 2 k-chunks (scores already in log2 domain) ----
    v8f c[4];
    #pragma unroll
    for (int t = 0; t < 4; ++t) {
      const int n = t * 16 + lq;              // key index in tile = B column
      const _Float16* krow = &sK[n * D_SZ];
      v16h b0, b1;
      #pragma unroll
      for (int e = 0; e < 16; ++e) b0[e] = krow[hi * 16 + e];        // d = 0..31
      #pragma unroll
      for (int e = 0; e < 16; ++e) b1[e] = krow[32 + hi * 16 + e];   // d = 32..63
      v8f z = vzero;
      z = __builtin_amdgcn_wmma_f32_16x16x32_f16(false, aQ[0], false, b0,
                                                 (short)0, z, false, false);
      z = __builtin_amdgcn_wmma_f32_16x16x32_f16(false, aQ[1], false, b1,
                                                 (short)0, z, false, false);
      c[t] = z;
    }

    // ---- online softmax in exp2 domain (fused-DPP row reductions) ----
    #pragma unroll
    for (int r = 0; r < 8; ++r) {
      float tmax = fmaxf(fmaxf(c[0][r], c[1][r]), fmaxf(c[2][r], c[3][r]));
      tmax = red_max16(tmax);
      const float newm  = fmaxf(mrow[r], tmax);
      const float alpha = __builtin_amdgcn_exp2f(mrow[r] - newm);
      float rs = 0.0f;
      #pragma unroll
      for (int t = 0; t < 4; ++t) {
        float pv = __builtin_amdgcn_exp2f(c[t][r] - newm);
        c[t][r] = pv;                       // reuse c[] as P
        rs += pv;
      }
      rs = red_sum16(rs);
      lrow[r] = lrow[r] * alpha + rs;
      mrow[r] = newm;
      #pragma unroll
      for (int t = 0; t < 4; ++t) acc[t][r] *= alpha;
    }

    // ---- reshape P: C/D layout -> LDS strip -> A layout ----
    _Float16* ps = sP[wave];
    #pragma unroll
    for (int r = 0; r < 8; ++r) {
      const int m = r + 8 * hi;
      #pragma unroll
      for (int t = 0; t < 4; ++t)
        ps[m * BN + t * 16 + lq] = (_Float16)c[t][r];
    }
    // same-wave LDS ops are in-order (ISA ch.11); sP is wave-private.
    v16h aP[2];
    {
      const _Float16* prow = &ps[lq * BN];
      #pragma unroll
      for (int kc = 0; kc < 2; ++kc) {
        const int kb = kc * 32;
        #pragma unroll
        for (int e = 0; e < 8; ++e) aP[kc][e]     = prow[kb + hi * 16 + e];
        #pragma unroll
        for (int e = 0; e < 8; ++e) aP[kc][8 + e] = prow[kb + hi * 16 + 8 + e];
      }
    }

    // ---- O += P V : 4 dv-tiles x 2 k-chunks ----
    #pragma unroll
    for (int t = 0; t < 4; ++t) {
      const int dv = t * 16 + lq;             // output column = B column
      const _Float16* vcol = &sVt[dv * BN];
      v16h b0, b1;
      #pragma unroll
      for (int e = 0; e < 16; ++e) b0[e] = vcol[hi * 16 + e];        // k = 0..31
      #pragma unroll
      for (int e = 0; e < 16; ++e) b1[e] = vcol[32 + hi * 16 + e];   // k = 32..63
      acc[t] = __builtin_amdgcn_wmma_f32_16x16x32_f16(false, aP[0], false, b0,
                                                      (short)0, acc[t], false, false);
      acc[t] = __builtin_amdgcn_wmma_f32_16x16x32_f16(false, aP[1], false, b1,
                                                      (short)0, acc[t], false, false);
    }
  }

  // ---- epilogue: normalize by l and store fp32 ----
  float* Og = O + ((size_t)batch * N_SZ + (size_t)qblk * BM + wave * 16) * D_SZ;
  #pragma unroll
  for (int r = 0; r < 8; ++r) {
    const int m = r + 8 * hi;
    const float inv = 1.0f / lrow[r];
    #pragma unroll
    for (int t = 0; t < 4; ++t)
      Og[m * D_SZ + t * 16 + lq] = acc[t][r] * inv;
  }
}

extern "C" void kernel_launch(void* const* d_in, const int* in_sizes, int n_in,
                              void* d_out, int out_size, void* d_ws, size_t ws_size,
                              hipStream_t stream) {
  const float* q = (const float*)d_in[0];
  const float* k = (const float*)d_in[1];
  const float* v = (const float*)d_in[2];
  float* o = (float*)d_out;
  dim3 grid(B_SZ * (N_SZ / BM));   // 16 * 16 = 256 workgroups
  dim3 block(NTHREADS);            // 8 wave32
  hipLaunchKernelGGL(fa_wmma_kernel, grid, block, 0, stream, q, k, v, o);
}
